// DCNBlock_64630667870672
// MI455X (gfx1250) — matrix operand order repaired and checked by reading, not compiled, not run on hardware
//
#include <hip/hip_runtime.h>
#include <hip/hip_bf16.h>
#include <math.h>

// ---------------------------------------------------------------- types
typedef __bf16 bf16;
typedef __attribute__((ext_vector_type(16))) __bf16 bfx16;
typedef __attribute__((ext_vector_type(8)))  float  fx8;
typedef __attribute__((ext_vector_type(4)))  unsigned int u32x4;

union FragBF { bfx16 v; u32x4 q[2]; };

// ---------------------------------------------------------------- shapes
static constexpr int cB   = 8;
static constexpr int cC   = 256;
static constexpr int cH   = 56;
static constexpr int cW   = 56;
static constexpr int cHW  = cH * cW;          // 3136
static constexpr int cL   = cB * cHW;         // 25088
static constexpr int cHID = 1024;
static constexpr int cG   = 16;               // deformable groups
static constexpr int cP   = 9;                // taps
static constexpr int cNOM = 448;              // 288 off + 144 mask + 16 pad
static constexpr float cEPS = 1e-5f;

__device__ __forceinline__ float gelu_f(float x) {
    return 0.5f * x * (1.0f + erff(x * 0.70710678118654752440f));
}

// ================================================================ WMMA GEMM
// A: bf16 [M x K] row-major. Bt: bf16 [N x K] row-major (pre-transposed weight).
// Block tile 128x64, K-step 32, 8 waves (wave32): wave grid 4(M) x 2(N),
// each wave computes a 32x32 patch = 2x2 WMMA tiles (4 accumulators),
// so per K-step: 8 ds_load_b128 feed 4 WMMAs (2 loads/WMMA vs 3 before).
// Double-buffered LDS: next K-tile staged to registers during compute,
// single __syncthreads per K-step.
#define BM 128
#define BN 64
#define BK 32
#define BKP 40   // padded LDS K-stride (80B: 16B-aligned, conflict-free)

// MODE 0: outF = acc + bias                         (f32, ld=N)
// MODE 1: outF = acc + bias + xres(NCHW)            (f32 NHWC, N==256)
// MODE 2: outB = bf16( gelu(acc)*ep0[n] + ep1[n] )  (folded BN)
// MODE 3: outNCHW = acc + ep0(NHWC resid)           (final store, N==256)
template <int MODE>
__global__ __launch_bounds__(256) void k_gemm(
    const bf16* __restrict__ A, const bf16* __restrict__ Bt,
    const float* __restrict__ bias,
    float* __restrict__ outF, bf16* __restrict__ outB,
    const float* __restrict__ ep0, const float* __restrict__ ep1,
    int M, int N, int K)
{
    __shared__ __align__(16) bf16 lA[2][BM][BKP];   // 2 x 10240 B
    __shared__ __align__(16) bf16 lB[2][BN][BKP];   // 2 x  5120 B

    const int tid  = threadIdx.x;
    const int m0   = blockIdx.y * BM;
    const int n0   = blockIdx.x * BN;
    const int wave = tid >> 5;
    const int lane = tid & 31;
    const int wm   = wave >> 1;       // 0..3  (M)
    const int wn   = wave & 1;        // 0..1  (N)
    const int lr   = lane & 15;
    const int lh   = lane >> 4;

    const int rowL = tid >> 2;        // 0..63
    const int colL = (tid & 3) << 3;  // 0,8,16,24 : 8 bf16 per thread

    fx8 acc00 = {}, acc01 = {}, acc10 = {}, acc11 = {};

    const bf16* gA0 = A  + (size_t)(m0 + rowL) * K + colL;        // rows 0..63
    const bf16* gA1 = A  + (size_t)(m0 + 64 + rowL) * K + colL;   // rows 64..127
    const bf16* gB  = Bt + (size_t)(n0 + rowL) * K + colL;

    const int ktiles = K / BK;
    u32x4 ra0 = *(const u32x4*)gA0;
    u32x4 ra1 = *(const u32x4*)gA1;
    u32x4 rb  = *(const u32x4*)gB;

    int cur = 0;
    for (int kt = 0; kt < ktiles; ++kt) {
        // commit staged tile to the current LDS buffer
        *(u32x4*)&lA[cur][rowL][colL]      = ra0;
        *(u32x4*)&lA[cur][64 + rowL][colL] = ra1;
        *(u32x4*)&lB[cur][rowL][colL]      = rb;
        __syncthreads();

        // stage next K-tile into registers while this one is consumed
        if (kt + 1 < ktiles) {
            const int off = (kt + 1) * BK;
            ra0 = *(const u32x4*)(gA0 + off);
            ra1 = *(const u32x4*)(gA1 + off);
            rb  = *(const u32x4*)(gB + off);
            if (kt + 2 < ktiles) {               // global_prefetch_b8
                __builtin_prefetch(gA0 + off + BK, 0, 1);
                __builtin_prefetch(gA1 + off + BK, 0, 1);
                __builtin_prefetch(gB + off + BK, 0, 1);
            }
        }

        // fragments: A rows wm*32 + t*16 + lr; B cols wn*32 + u*16 + lr
        FragBF fa0, fa1, fb0, fb1;
        fa0.q[0] = *(const u32x4*)&lA[cur][wm * 32 + lr][lh * 8];
        fa0.q[1] = *(const u32x4*)&lA[cur][wm * 32 + lr][16 + lh * 8];
        fa1.q[0] = *(const u32x4*)&lA[cur][wm * 32 + 16 + lr][lh * 8];
        fa1.q[1] = *(const u32x4*)&lA[cur][wm * 32 + 16 + lr][16 + lh * 8];
        fb0.q[0] = *(const u32x4*)&lB[cur][wn * 32 + lr][lh * 8];
        fb0.q[1] = *(const u32x4*)&lB[cur][wn * 32 + lr][16 + lh * 8];
        fb1.q[0] = *(const u32x4*)&lB[cur][wn * 32 + 16 + lr][lh * 8];
        fb1.q[1] = *(const u32x4*)&lB[cur][wn * 32 + 16 + lr][16 + lh * 8];

        acc00 = __builtin_amdgcn_wmma_f32_16x16x32_bf16(
            false, fa0.v, false, fb0.v, (short)0, acc00, false, false);
        acc01 = __builtin_amdgcn_wmma_f32_16x16x32_bf16(
            false, fa0.v, false, fb1.v, (short)0, acc01, false, false);
        acc10 = __builtin_amdgcn_wmma_f32_16x16x32_bf16(
            false, fa1.v, false, fb0.v, (short)0, acc10, false, false);
        acc11 = __builtin_amdgcn_wmma_f32_16x16x32_bf16(
            false, fa1.v, false, fb1.v, (short)0, acc11, false, false);
        cur ^= 1;
    }

#pragma unroll
    for (int t = 0; t < 2; ++t) {
#pragma unroll
        for (int u = 0; u < 2; ++u) {
            const fx8 a = t ? (u ? acc11 : acc10) : (u ? acc01 : acc00);
            const int col   = n0 + wn * 32 + u * 16 + lr;
            const int rbase = m0 + wm * 32 + t * 16 + lh * 8;
#pragma unroll
            for (int j = 0; j < 8; ++j) {
                const int row = rbase + j;
                float v = a[j];
                if (MODE == 0) {
                    outF[(size_t)row * N + col] = v + bias[col];
                } else if (MODE == 1) {
                    const int b  = row / cHW;
                    const int hw = row - b * cHW;
                    v += bias[col] + ep0[((size_t)(b * cC + col)) * cHW + hw];
                    outF[(size_t)row * N + col] = v;
                } else if (MODE == 2) {
                    float tg = gelu_f(v);
                    outB[(size_t)row * N + col] = (bf16)(tg * ep0[col] + ep1[col]);
                } else { // MODE 3
                    const int b  = row / cHW;
                    const int hw = row - b * cHW;
                    outF[((size_t)(b * cC + col)) * cHW + hw] =
                        v + ep0[(size_t)row * N + col];
                }
            }
        }
    }
}

// ================================================================ reductions
__global__ __launch_bounds__(256) void k_reduce_partial(
    const float* __restrict__ x, float* __restrict__ part,
    int perSample, int chunk)
{
    const int s = blockIdx.y;
    const int start = blockIdx.x * chunk;
    int end = start + chunk;
    if (end > perSample) end = perSample;
    const float* base = x + (size_t)s * perSample;
    float s1 = 0.f, s2 = 0.f;
    for (int i = start + threadIdx.x; i < end; i += 256) {
        float v = base[i];
        s1 += v; s2 += v * v;
    }
    __shared__ float sh1[256], sh2[256];
    sh1[threadIdx.x] = s1; sh2[threadIdx.x] = s2;
    __syncthreads();
    for (int off = 128; off > 0; off >>= 1) {
        if (threadIdx.x < off) {
            sh1[threadIdx.x] += sh1[threadIdx.x + off];
            sh2[threadIdx.x] += sh2[threadIdx.x + off];
        }
        __syncthreads();
    }
    if (threadIdx.x == 0) {
        part[(size_t)(s * gridDim.x + blockIdx.x) * 2 + 0] = sh1[0];
        part[(size_t)(s * gridDim.x + blockIdx.x) * 2 + 1] = sh2[0];
    }
}

__global__ __launch_bounds__(256) void k_reduce_final(
    const float* __restrict__ part, float* __restrict__ stats,
    int nb, float invN)
{
    const int s = blockIdx.x;
    float s1 = 0.f, s2 = 0.f;
    for (int i = threadIdx.x; i < nb; i += 256) {
        s1 += part[(size_t)(s * nb + i) * 2 + 0];
        s2 += part[(size_t)(s * nb + i) * 2 + 1];
    }
    __shared__ float sh1[256], sh2[256];
    sh1[threadIdx.x] = s1; sh2[threadIdx.x] = s2;
    __syncthreads();
    for (int off = 128; off > 0; off >>= 1) {
        if (threadIdx.x < off) {
            sh1[threadIdx.x] += sh1[threadIdx.x + off];
            sh2[threadIdx.x] += sh2[threadIdx.x + off];
        }
        __syncthreads();
    }
    if (threadIdx.x == 0) {
        float mean = sh1[0] * invN;
        float var  = sh2[0] * invN - mean * mean;
        stats[s * 2 + 0] = mean;
        stats[s * 2 + 1] = rsqrtf(var + cEPS);
    }
}

// ================================================================ GN apply
// x NCHW -> normalized NHWC (f32 + bf16)
__global__ __launch_bounds__(256) void k_gn_x(
    const float* __restrict__ x, const float* __restrict__ stats,
    const float* __restrict__ g, const float* __restrict__ bb,
    float* __restrict__ outF, bf16* __restrict__ outB)
{
    const int i = blockIdx.x * 256 + threadIdx.x;
    if (i >= cL * cC) return;
    const int c = i & (cC - 1);
    const int m = i >> 8;
    const int b = m / cHW;
    const int hw = m - b * cHW;
    const float v = x[((size_t)(b * cC + c)) * cHW + hw];
    const float r = (v - stats[b * 2]) * stats[b * 2 + 1] * g[c] + bb[c];
    outF[i] = r;
    outB[i] = (bf16)r;
}

// src NHWC f32 -> GN (+ optional GELU) -> bf16 NHWC
__global__ __launch_bounds__(256) void k_gn_nhwc(
    const float* __restrict__ src, const float* __restrict__ stats,
    const float* __restrict__ g, const float* __restrict__ bb,
    bf16* __restrict__ outB, int doGelu)
{
    const int i = blockIdx.x * 256 + threadIdx.x;
    if (i >= cL * cC) return;
    const int c = i & (cC - 1);
    const int m = i >> 8;
    const int b = m / cHW;
    float r = (src[i] - stats[b * 2]) * stats[b * 2 + 1] * g[c] + bb[c];
    if (doGelu) r = gelu_f(r);
    outB[i] = (bf16)r;
}

// ================================================================ depthwise 3x3 (NHWC)
__global__ __launch_bounds__(256) void k_dwconv(
    const float* __restrict__ xn, const float* __restrict__ w,
    const float* __restrict__ bias, float* __restrict__ out)
{
    const int i = blockIdx.x * 256 + threadIdx.x;
    if (i >= cL * cC) return;
    const int c = i & (cC - 1);
    const int m = i >> 8;
    const int b = m / cHW;
    const int hw = m - b * cHW;
    const int hh = hw / cW;
    const int ww = hw - hh * cW;
    float acc = bias[c];
#pragma unroll
    for (int ky = 0; ky < 3; ++ky) {
        const int yy = hh + ky - 1;
        if (yy < 0 || yy >= cH) continue;
#pragma unroll
        for (int kx = 0; kx < 3; ++kx) {
            const int xx = ww + kx - 1;
            if (xx < 0 || xx >= cW) continue;
            acc += xn[(((size_t)(b * cHW + yy * cW + xx)) << 8) + c] *
                   w[c * 9 + ky * 3 + kx];
        }
    }
    out[i] = acc;
}

// ================================================================ softmax over 9 taps
__global__ __launch_bounds__(256) void k_softmax(float* __restrict__ om)
{
    const int t = blockIdx.x * 256 + threadIdx.x;
    if (t >= cL * cG) return;
    const int m = t >> 4;
    const int g = t & 15;
    float* p = om + (size_t)m * cNOM + 288 + g * cP;
    float mx = p[0];
#pragma unroll
    for (int k = 1; k < cP; ++k) mx = fmaxf(mx, p[k]);
    float s = 0.f;
    float e[cP];
#pragma unroll
    for (int k = 0; k < cP; ++k) { e[k] = __expf(p[k] - mx); s += e[k]; }
    const float inv = 1.f / s;
#pragma unroll
    for (int k = 0; k < cP; ++k) p[k] = e[k] * inv;
}

// ================================================================ deformable sampling
// one block per spatial point; 256 threads = channels; offsets/masks in LDS
__global__ __launch_bounds__(256) void k_dcn(
    const float* __restrict__ xproj, const float* __restrict__ om,
    bf16* __restrict__ out)
{
    __shared__ float soff[cG * cP * 2];   // 288
    __shared__ float smask[cG * cP];      // 144
    const int m = blockIdx.x;
    const float* row = om + (size_t)m * cNOM;
    for (int i = threadIdx.x; i < 432; i += 256) {
        if (i < 288) soff[i] = row[i];
        else         smask[i - 288] = row[i];
    }
    __syncthreads();

    const int c  = threadIdx.x;
    const int g  = c >> 4;
    const int b  = m / cHW;
    const int hw = m - b * cHW;
    const int hh = hw / cW;
    const int ww = hw - hh * cW;
    const float* img = xproj + (((size_t)b) * cHW << 8);

    float acc = 0.f;
#pragma unroll
    for (int p = 0; p < cP; ++p) {
        const float dxp = (float)(p / 3 - 1);   // meshgrid(ij): x uses row idx
        const float dyp = (float)(p % 3 - 1);
        const float sx = (float)ww + dxp + soff[(g * cP + p) * 2 + 0];
        const float sy = (float)hh + dyp + soff[(g * cP + p) * 2 + 1];
        const float x0f = floorf(sx), y0f = floorf(sy);
        const int x0 = (int)x0f, y0 = (int)y0f;
        const float fx = sx - x0f, fy = sy - y0f;
        float v00 = 0.f, v10 = 0.f, v01 = 0.f, v11 = 0.f;
        const bool xa = (x0 >= 0 && x0 < cW),  xb = (x0 + 1 >= 0 && x0 + 1 < cW);
        const bool ya = (y0 >= 0 && y0 < cH),  yb = (y0 + 1 >= 0 && y0 + 1 < cH);
        if (xa && ya) v00 = img[(((size_t)(y0 * cW + x0)) << 8) + c];
        if (xb && ya) v10 = img[(((size_t)(y0 * cW + x0 + 1)) << 8) + c];
        if (xa && yb) v01 = img[(((size_t)((y0 + 1) * cW + x0)) << 8) + c];
        if (xb && yb) v11 = img[(((size_t)((y0 + 1) * cW + x0 + 1)) << 8) + c];
        const float bil = v00 * (1.f - fx) * (1.f - fy) + v10 * fx * (1.f - fy)
                        + v01 * (1.f - fx) * fy         + v11 * fx * fy;
        acc += smask[g * cP + p] * bil;
    }
    out[(size_t)m * cC + c] = (bf16)acc;
}

// ================================================================ multi-scale dw conv
// + residual + GELU + folded BN2 -> bf16
__global__ __launch_bounds__(256) void k_msconv(
    const bf16* __restrict__ x1,
    const float* __restrict__ w0, const float* __restrict__ b0,
    const float* __restrict__ w1, const float* __restrict__ b1,
    const float* __restrict__ w2, const float* __restrict__ b2,
    const float* __restrict__ w3, const float* __restrict__ b3,
    const float* __restrict__ bn2s, const float* __restrict__ bn2t,
    bf16* __restrict__ out)
{
    const int i = blockIdx.x * 256 + threadIdx.x;
    if (i >= cL * cHID) return;
    const int c  = i & (cHID - 1);
    const int m  = i >> 10;
    const int sc = c >> 8;           // 0..3 -> k = 1,3,5,7
    const int cw = c & 255;
    const int ksz  = 2 * sc + 1;
    const int half = sc;
    const float* w; const float* bi;
    if      (sc == 0) { w = w0; bi = b0; }
    else if (sc == 1) { w = w1; bi = b1; }
    else if (sc == 2) { w = w2; bi = b2; }
    else              { w = w3; bi = b3; }

    const int b  = m / cHW;
    const int hw = m - b * cHW;
    const int hh = hw / cW;
    const int ww = hw - hh * cW;

    float acc = bi[cw];
    for (int ky = 0; ky < ksz; ++ky) {
        const int yy = hh + ky - half;
        if (yy < 0 || yy >= cH) continue;
        for (int kx = 0; kx < ksz; ++kx) {
            const int xx = ww + kx - half;
            if (xx < 0 || xx >= cW) continue;
            acc += (float)x1[(((size_t)(b * cHW + yy * cW + xx)) << 10) + c] *
                   w[cw * ksz * ksz + ky * ksz + kx];
        }
    }
    const float t = gelu_f(acc + (float)x1[i]);
    out[i] = (bf16)(t * bn2s[c] + bn2t[c]);
}

// ================================================================ weight prep
// f32 [K x N] -> bf16 [Npad x K] transposed (zero-fill rows n >= N)
__global__ __launch_bounds__(256) void k_transposeW(
    const float* __restrict__ src, bf16* __restrict__ dst,
    int K, int N, int Npad)
{
    const int t = blockIdx.x * 256 + threadIdx.x;
    if (t >= Npad * K) return;
    const int n = t / K;
    const int k = t - n * K;
    dst[t] = (n < N) ? (bf16)src[(size_t)k * N + n] : (bf16)0.f;
}

__global__ __launch_bounds__(256) void k_convertW(
    const float* __restrict__ src, bf16* __restrict__ dst, int cnt)
{
    const int t = blockIdx.x * 256 + threadIdx.x;
    if (t < cnt) dst[t] = (bf16)src[t];
}

__global__ __launch_bounds__(256) void k_biasom(
    const float* __restrict__ offb, const float* __restrict__ maskb,
    float* __restrict__ dst)
{
    const int t = blockIdx.x * 256 + threadIdx.x;
    if (t >= cNOM) return;
    float v = 0.f;
    if (t < 288)      v = offb[t];
    else if (t < 432) v = maskb[t - 288];
    dst[t] = v;
}

__global__ __launch_bounds__(256) void k_bnfold(
    const float* __restrict__ g, const float* __restrict__ b,
    const float* __restrict__ m, const float* __restrict__ v,
    float* __restrict__ s, float* __restrict__ t, int n)
{
    const int i = blockIdx.x * 256 + threadIdx.x;
    if (i >= n) return;
    const float sc = g[i] * rsqrtf(v[i] + cEPS);
    s[i] = sc;
    t[i] = b[i] - m[i] * sc;
}

// ================================================================ host
extern "C" void kernel_launch(void* const* d_in, const int* in_sizes, int n_in,
                              void* d_out, int out_size, void* d_ws, size_t ws_size,
                              hipStream_t stream) {
    (void)in_sizes; (void)n_in; (void)out_size; (void)ws_size;
    // input order: x, then params in dict-insertion order
    const float* x      = (const float*)d_in[0];
    const float* n1_g   = (const float*)d_in[1];
    const float* n1_b   = (const float*)d_in[2];
    const float* n2_g   = (const float*)d_in[3];
    const float* n2_b   = (const float*)d_in[4];
    const float* dw_w   = (const float*)d_in[5];
    const float* dw_b   = (const float*)d_in[6];
    const float* dwn_g  = (const float*)d_in[7];
    const float* dwn_b  = (const float*)d_in[8];
    const float* off_w  = (const float*)d_in[9];
    const float* off_b  = (const float*)d_in[10];
    const float* mask_w = (const float*)d_in[11];
    const float* mask_b = (const float*)d_in[12];
    const float* in_w   = (const float*)d_in[13];
    const float* in_b   = (const float*)d_in[14];
    const float* out_w  = (const float*)d_in[15];
    const float* out_b  = (const float*)d_in[16];
    const float* fc1_w  = (const float*)d_in[17];
    const float* bn1_g  = (const float*)d_in[18];
    const float* bn1_b  = (const float*)d_in[19];
    const float* bn1_m  = (const float*)d_in[20];
    const float* bn1_v  = (const float*)d_in[21];
    const float* bn2_g  = (const float*)d_in[22];
    const float* bn2_b  = (const float*)d_in[23];
    const float* bn2_m  = (const float*)d_in[24];
    const float* bn2_v  = (const float*)d_in[25];
    const float* fc2_w  = (const float*)d_in[26];
    const float* ms_w0  = (const float*)d_in[27];
    const float* ms_b0  = (const float*)d_in[28];
    const float* ms_w1  = (const float*)d_in[29];
    const float* ms_b1  = (const float*)d_in[30];
    const float* ms_w2  = (const float*)d_in[31];
    const float* ms_b2  = (const float*)d_in[32];
    const float* ms_w3  = (const float*)d_in[33];
    const float* ms_b3  = (const float*)d_in[34];

    // ---- workspace carve
    char* wp = (char*)d_ws;
    auto carve = [&](size_t bytes) -> void* {
        void* r = (void*)wp;
        wp += (bytes + 255) & ~(size_t)255;
        return r;
    };
    bf16*  bt_in   = (bf16*)carve((size_t)cC * cC * 2);
    bf16*  bt_out  = (bf16*)carve((size_t)cC * cC * 2);
    bf16*  bt_om   = (bf16*)carve((size_t)cNOM * cC * 2);
    bf16*  bt_fc1  = (bf16*)carve((size_t)cHID * cC * 2);
    bf16*  bt_fc2  = (bf16*)carve((size_t)cC * cHID * 2);
    float* bias_om = (float*)carve(cNOM * 4);
    float* bn1s    = (float*)carve(cHID * 4);
    float* bn1t    = (float*)carve(cHID * 4);
    float* bn2s    = (float*)carve(cHID * 4);
    float* bn2t    = (float*)carve(cHID * 4);
    const int NB = 112;
    float* part    = (float*)carve((size_t)cB * NB * 2 * 4);
    float* stats_x = (float*)carve(cB * 2 * 4);
    float* stats_d = (float*)carve(cB * 2 * 4);
    float* stats_h = (float*)carve(cB * 2 * 4);
    float* xn_f    = (float*)carve((size_t)cL * cC * 4);
    bf16*  xn_b    = (bf16*)carve((size_t)cL * cC * 2);
    float* xproj   = (float*)carve((size_t)cL * cC * 4);
    float* d_f     = (float*)carve((size_t)cL * cC * 4);
    bf16*  d_bf    = (bf16*)carve((size_t)cL * cC * 2);
    float* om      = (float*)carve((size_t)cL * cNOM * 4);
    bf16*  dcn_b   = (bf16*)carve((size_t)cL * cC * 2);
    float* h_f     = (float*)carve((size_t)cL * cC * 4);
    bf16*  hn_b    = (bf16*)carve((size_t)cL * cC * 2);
    bf16*  x1_b    = (bf16*)carve((size_t)cL * cHID * 2);
    bf16*  a2_b    = (bf16*)carve((size_t)cL * cHID * 2);
    float* outp    = (float*)d_out;

    auto blks = [](long n) { return (int)((n + 255) / 256); };

    // ---- weight prep (bf16, N x K transposed)
    k_transposeW<<<blks((long)cC * cC), 256, 0, stream>>>(in_w,  bt_in,  cC, cC, cC);
    k_transposeW<<<blks((long)cC * cC), 256, 0, stream>>>(out_w, bt_out, cC, cC, cC);
    k_transposeW<<<blks((long)288 * cC), 256, 0, stream>>>(off_w,  bt_om,            cC, 288, 288);
    k_transposeW<<<blks((long)144 * cC), 256, 0, stream>>>(mask_w, bt_om + 288 * cC, cC, 144, 144);
    k_transposeW<<<blks((long)16 * cC),  256, 0, stream>>>(mask_w, bt_om + 432 * cC, cC, 0,   16);
    k_convertW<<<blks((long)cHID * cC), 256, 0, stream>>>(fc1_w, bt_fc1, cHID * cC); // already N x K
    k_convertW<<<blks((long)cC * cHID), 256, 0, stream>>>(fc2_w, bt_fc2, cC * cHID); // already N x K
    k_biasom<<<2, 256, 0, stream>>>(off_b, mask_b, bias_om);
    k_bnfold<<<blks(cHID), 256, 0, stream>>>(bn1_g, bn1_b, bn1_m, bn1_v, bn1s, bn1t, cHID);
    k_bnfold<<<blks(cHID), 256, 0, stream>>>(bn2_g, bn2_b, bn2_m, bn2_v, bn2s, bn2t, cHID);

    const int perS  = cHW * cC;                 // 802816 per sample
    const int chunk = (perS + NB - 1) / NB;
    const float invN = 1.0f / (float)perS;
    dim3 rp(NB, cB);

    // ---- gn1(x) -> xn (NHWC f32 + bf16)
    k_reduce_partial<<<rp, 256, 0, stream>>>(x, part, perS, chunk);
    k_reduce_final<<<cB, 256, 0, stream>>>(part, stats_x, NB, invN);
    k_gn_x<<<blks((long)cL * cC), 256, 0, stream>>>(x, stats_x, n1_g, n1_b, xn_f, xn_b);

    // ---- in_proj GEMM: xproj = xn @ in_w + in_b
    dim3 g256(cC / BN, cL / BM);
    k_gemm<0><<<g256, 256, 0, stream>>>(xn_b, bt_in, in_b, xproj, nullptr,
                                        nullptr, nullptr, cL, cC, cC);

    // ---- depthwise 3x3 + GN + GELU -> d_bf
    k_dwconv<<<blks((long)cL * cC), 256, 0, stream>>>(xn_f, dw_w, dw_b, d_f);
    k_reduce_partial<<<rp, 256, 0, stream>>>(d_f, part, perS, chunk);
    k_reduce_final<<<cB, 256, 0, stream>>>(part, stats_d, NB, invN);
    k_gn_nhwc<<<blks((long)cL * cC), 256, 0, stream>>>(d_f, stats_d, dwn_g, dwn_b, d_bf, 1);

    // ---- offset+mask GEMM (N padded to 448) + softmax on mask cols
    dim3 gom(cNOM / BN, cL / BM);
    k_gemm<0><<<gom, 256, 0, stream>>>(d_bf, bt_om, bias_om, om, nullptr,
                                       nullptr, nullptr, cL, cNOM, cC);
    k_softmax<<<blks((long)cL * cG), 256, 0, stream>>>(om);

    // ---- deformable sampling -> dcn_b (bf16 NHWC)
    k_dcn<<<cL, 256, 0, stream>>>(xproj, om, dcn_b);

    // ---- out_proj GEMM + bias + residual x (NCHW) -> h (f32 NHWC)
    k_gemm<1><<<g256, 256, 0, stream>>>(dcn_b, bt_out, out_b, h_f, nullptr,
                                        x, nullptr, cL, cC, cC);

    // ---- gn2(h) -> hn_b
    k_reduce_partial<<<rp, 256, 0, stream>>>(h_f, part, perS, chunk);
    k_reduce_final<<<cB, 256, 0, stream>>>(part, stats_h, NB, invN);
    k_gn_nhwc<<<blks((long)cL * cC), 256, 0, stream>>>(h_f, stats_h, n2_g, n2_b, hn_b, 0);

    // ---- fc1 GEMM + GELU + BN1 -> x1 (bf16)
    dim3 gfc1(cHID / BN, cL / BM);
    k_gemm<2><<<gfc1, 256, 0, stream>>>(hn_b, bt_fc1, nullptr, nullptr, x1_b,
                                        bn1s, bn1t, cL, cHID, cC);

    // ---- multi-scale depthwise + residual + GELU + BN2 -> a2 (bf16)
    k_msconv<<<blks((long)cL * cHID), 256, 0, stream>>>(
        x1_b, ms_w0, ms_b0, ms_w1, ms_b1, ms_w2, ms_b2, ms_w3, ms_b3,
        bn2s, bn2t, a2_b);

    // ---- fc2 GEMM + residual h -> final output (NCHW f32)
    k_gemm<3><<<g256, 256, 0, stream>>>(a2_b, bt_fc2, nullptr, outp, nullptr,
                                        h_f, nullptr, cL, cC, cHID);
}
